// AttentionPooling_50809463112055
// MI455X (gfx1250) — compile-verified, matrix operand order checked by reading, not run on hardware
//
#include <hip/hip_runtime.h>

// AttentionPooling (MIL) fused kernel for gfx1250 (MI455X).
// bags = 2048, BAG = 128 rows/bag, D = H = 1024.
// Prep kernel: W1 fp32 [K][N] -> bf16 transposed [N][K] in d_ws (2 MB, L2-resident).
// Main kernel: one workgroup (256 threads = 8 wave32) per bag:
//   A (bag's x tile, 128x1024) staged once to LDS as bf16 (258 KB),
//   B (W1t panels) double-buffered in LDS, 1 barrier/chunk, fetch overlapped,
//   v_wmma_f32_16x16x32_bf16 GEMM fused with tanh/W2 score, softmax, weighted sum.

#define NBAGS 2048
#define BAG   128
#define DIM   1024
#define HID   1024

#define LDA 1032      // DIM + 8 bf16 pad (break 64-bank alignment)
#define KC  128       // K chunk staged for W1t
#define NC  64        // N chunk (4 x 16 tiles)
#define LDB 136       // KC + 8 pad (row stride 272 B, 16B-aligned)

typedef __attribute__((ext_vector_type(16))) __bf16 bf16x16;
typedef __attribute__((ext_vector_type(8)))  __bf16 bf16x8;
typedef __attribute__((ext_vector_type(8)))  float  f32x8;

__device__ __forceinline__ unsigned short f2bf(float f) {
    // round-to-nearest-even fp32 -> bf16
    unsigned u = __float_as_uint(f);
    unsigned r = u + 0x7FFFu + ((u >> 16) & 1u);
    return (unsigned short)(r >> 16);
}

__device__ __forceinline__ float fast_tanh(float x) {
#if __has_builtin(__builtin_amdgcn_tanhf)
    return __builtin_amdgcn_tanhf(x);
#elif __has_builtin(__builtin_amdgcn_tanh_f32)
    return __builtin_amdgcn_tanh_f32(x);
#else
    // tanh(x) = 1 - 2/(exp(2x)+1); saturates correctly at +-inf
    float e = __expf(2.0f * x);
    return 1.0f - 2.0f / (e + 1.0f);
#endif
}

__device__ __forceinline__ bf16x16 ldsFrag(const unsigned short* p) {
    bf16x8 lo = *(const bf16x8*)(p);        // K 0..7 (or 8..15 for upper half-wave)
    bf16x8 hi = *(const bf16x8*)(p + 16);   // K +16
    return __builtin_shufflevector(lo, hi, 0,1,2,3,4,5,6,7,8,9,10,11,12,13,14,15);
}

// ---------------------------------------------------------------------------
// Prep: W1 [K=1024][N=1024] fp32  ->  W1t [N=1024][K=1024] bf16 (transposed)
// 64x64 tiles through LDS; coalesced reads and writes.
__global__ __launch_bounds__(256, 4)
void w1_to_bf16t_kernel(const float* __restrict__ W1, unsigned short* __restrict__ W1t)
{
    __shared__ unsigned short tile[64][72];   // pad 8 -> 144 B row stride (16B multiple)
    const int tid = threadIdx.x;
    const int kb  = (blockIdx.x & 15) * 64;
    const int nb  = (blockIdx.x >> 4) * 64;

    const int n  = tid & 63;                  // coalesced along N
    const int kr = tid >> 6;                  // 4 K-rows per sweep
    #pragma unroll 4
    for (int s = 0; s < 16; ++s) {
        const int k = kr + s * 4;
        tile[n][k] = f2bf(W1[(size_t)(kb + k) * HID + nb + n]);
    }
    __syncthreads();

    const int row = tid >> 2;                 // 0..63 (N-local)
    const int cq  = (tid & 3) * 2;            // two 16B chunks per thread
    #pragma unroll
    for (int j = 0; j < 2; ++j) {
        uint4 v = *(const uint4*)&tile[row][(cq + j) * 8];
        *(uint4*)(W1t + (size_t)(nb + row) * DIM + kb + (cq + j) * 8) = v;
    }
}

// ---------------------------------------------------------------------------
__global__ __launch_bounds__(256, 1)
void attn_pool_bag_kernel(const float* __restrict__ x,
                          const unsigned short* __restrict__ W1t,
                          const float* __restrict__ b1,
                          const float* __restrict__ W2,
                          const float* __restrict__ b2,
                          float* __restrict__ out)
{
    extern __shared__ char smem[];
    unsigned short* Alds = (unsigned short*)smem;          // [BAG][LDA] bf16 x-tile
    unsigned short* Bbuf = Alds + BAG * LDA;               // 2 x [NC][LDB] bf16 panels
    float* sc  = (float*)(Bbuf + 2 * NC * LDB);            // [BAG] scores
    float* wts = sc + BAG;                                 // [BAG] softmax weights
    float* red = wts + BAG;                                // [16]  reduction scratch

    const int tid   = threadIdx.x;
    const int lane  = tid & 31;
    const int wv    = tid >> 5;            // wave 0..7 -> M-tile
    const int half  = (lane >> 4) & 1;     // half-wave K offset 0/8 (A), rows 0-7/8-15 (C)
    const int lan16 = lane & 15;
    const int bag   = blockIdx.x;

    const size_t xbase = (size_t)bag * BAG * DIM;
    const int arow = 16 * wv + lan16;      // A fragment: lane -> row in this wave's M-tile

    // ---- B panel staging: pure bf16 copy, double-buffered, regs as pipe ----
    const int rowq = tid >> 4;             // 0..15  (16 threads per panel row)
    const int colq = tid & 15;             // 16B chunk within 256B row
    uint4 breg0, breg1, breg2, breg3;

    #define LOADB(n0_, k0_)                                                          \
        do {                                                                         \
            const unsigned short* s_ = W1t + (size_t)((n0_) + rowq) * DIM + (k0_) +  \
                                       colq * 8;                                     \
            breg0 = *(const uint4*)(s_);                                             \
            breg1 = *(const uint4*)(s_ + 16 * DIM);                                  \
            breg2 = *(const uint4*)(s_ + 32 * DIM);                                  \
            breg3 = *(const uint4*)(s_ + 48 * DIM);                                  \
        } while (0)

    #define STOREB(dst_)                                                             \
        do {                                                                         \
            unsigned short* d_ = (dst_) + rowq * LDB + colq * 8;                     \
            *(uint4*)(d_)            = breg0;                                        \
            *(uint4*)(d_ + 16 * LDB) = breg1;                                        \
            *(uint4*)(d_ + 32 * LDB) = breg2;                                        \
            *(uint4*)(d_ + 48 * LDB) = breg3;                                        \
        } while (0)

    LOADB(0, 0);   // first panel fetch overlaps with A staging below

    // ---- Phase 0: stage x[bag] -> LDS bf16, row major, padded ----------------
    {
        const int c = tid * 4;                       // 256 threads * 4 floats = 1024 cols
        for (int r = 0; r < BAG; ++r) {
            const float4 v = *(const float4*)(x + xbase + (size_t)r * DIM + c);
            unsigned long long p =
                 (unsigned long long)f2bf(v.x)
               | ((unsigned long long)f2bf(v.y) << 16)
               | ((unsigned long long)f2bf(v.z) << 32)
               | ((unsigned long long)f2bf(v.w) << 48);
            *(unsigned long long*)(Alds + r * LDA + c) = p;   // ds_store_b64
        }
    }

    const float b2v = b2[0];
    float scoreAcc[8];
    #pragma unroll
    for (int r = 0; r < 8; ++r) scoreAcc[r] = 0.f;

    // ---- Fused GEMM:  h = tanh(x@W1 + b1);  score += h @ W2 ------------------
    for (int nc = 0; nc < HID / NC; ++nc) {
        const int n0 = nc * NC;
        f32x8 acc0 = {0,0,0,0,0,0,0,0};
        f32x8 acc1 = acc0, acc2 = acc0, acc3 = acc0;

        for (int kc = 0; kc < DIM / KC; ++kc) {
            unsigned short* cur = Bbuf + (kc & 1) * (NC * LDB);
            // stage chunk kc (breg -> LDS); prior readers of `cur` finished
            // before the barrier of iteration kc-1 (double buffer)
            STOREB(cur);
            // issue next chunk's global loads; latency hidden behind barrier+WMMAs
            if (kc < DIM / KC - 1)      LOADB(n0, (kc + 1) * KC);
            else if (nc < HID / NC - 1) LOADB(n0 + NC, 0);
            __syncthreads();            // `cur` staged & visible (also covers A on iter 0)

            // preload all A fragments of this chunk, then 4 independent WMMAs/k-step
            bf16x16 af[4];
            #pragma unroll
            for (int ks = 0; ks < 4; ++ks)
                af[ks] = ldsFrag(Alds + arow * LDA + kc * KC + ks * 32 + half * 8);

            #pragma unroll
            for (int ks = 0; ks < 4; ++ks) {
                const int kl = ks * 32 + half * 8;
                bf16x16 bf0 = ldsFrag(cur + (     lan16) * LDB + kl);
                bf16x16 bf1 = ldsFrag(cur + (16 + lan16) * LDB + kl);
                bf16x16 bf2 = ldsFrag(cur + (32 + lan16) * LDB + kl);
                bf16x16 bf3 = ldsFrag(cur + (48 + lan16) * LDB + kl);
                acc0 = __builtin_amdgcn_wmma_f32_16x16x32_bf16(
                           false, af[ks], false, bf0, (short)0, acc0, false, false);
                acc1 = __builtin_amdgcn_wmma_f32_16x16x32_bf16(
                           false, af[ks], false, bf1, (short)0, acc1, false, false);
                acc2 = __builtin_amdgcn_wmma_f32_16x16x32_bf16(
                           false, af[ks], false, bf2, (short)0, acc2, false, false);
                acc3 = __builtin_amdgcn_wmma_f32_16x16x32_bf16(
                           false, af[ks], false, bf3, (short)0, acc3, false, false);
            }
        }

        // epilogue: fold bias + tanh + rank-1 W2 contraction into score partials
        #pragma unroll
        for (int t = 0; t < 4; ++t) {
            const int   n  = n0 + 16 * t + lan16;
            const float bb = b1[n];
            const float w2 = W2[n];
            const f32x8 a  = (t == 0) ? acc0 : (t == 1) ? acc1 : (t == 2) ? acc2 : acc3;
            #pragma unroll
            for (int r = 0; r < 8; ++r)
                scoreAcc[r] += fast_tanh(a[r] + bb) * w2;   // row = 16*wv + 8*half + r
        }
    }

    // ---- Reduce score partials across the 16 lanes sharing a row -------------
    #pragma unroll
    for (int r = 0; r < 8; ++r) {
        float v = scoreAcc[r];
        #pragma unroll
        for (int m = 1; m < 16; m <<= 1) v += __shfl_xor(v, m, 32);
        if (lan16 == 0) sc[16 * wv + 8 * half + r] = v + b2v;
    }
    __syncthreads();

    // ---- Stable softmax over the 128 scores ----------------------------------
    float s = (tid < BAG) ? sc[tid] : -3.402823466e38f;
    #pragma unroll
    for (int m = 16; m >= 1; m >>= 1) s = fmaxf(s, __shfl_xor(s, m, 32));
    if (lane == 0) red[wv] = s;
    __syncthreads();
    float mx = red[0];
    #pragma unroll
    for (int i = 1; i < 8; ++i) mx = fmaxf(mx, red[i]);

    float e = (tid < BAG) ? expf(sc[tid] - mx) : 0.f;
    float es = e;
    #pragma unroll
    for (int m = 16; m >= 1; m >>= 1) es += __shfl_xor(es, m, 32);
    if (lane == 0) red[8 + wv] = es;
    __syncthreads();
    float tot = red[8];
    #pragma unroll
    for (int i = 1; i < 8; ++i) tot += red[8 + i];
    if (tid < BAG) wts[tid] = e / tot;
    __syncthreads();

    // ---- Weighted bag sum: out[bag] = sum_i w_i * x[i]  (fp32 from global) ---
    {
        const int c = tid * 4;
        float o0 = 0.f, o1 = 0.f, o2 = 0.f, o3 = 0.f;
        for (int i = 0; i < BAG; ++i) {
            const float wi = wts[i];                       // LDS broadcast
            const float4 v = *(const float4*)(x + xbase + (size_t)i * DIM + c);
            o0 += wi * v.x; o1 += wi * v.y; o2 += wi * v.z; o3 += wi * v.w;
        }
        float4 o; o.x = o0; o.y = o1; o.z = o2; o.w = o3;
        *(float4*)(out + (size_t)bag * DIM + c) = o;
    }
    #undef LOADB
    #undef STOREB
}

extern "C" void kernel_launch(void* const* d_in, const int* in_sizes, int n_in,
                              void* d_out, int out_size, void* d_ws, size_t ws_size,
                              hipStream_t stream) {
    (void)in_sizes; (void)n_in; (void)out_size; (void)ws_size;
    const float* x  = (const float*)d_in[0];
    const float* W1 = (const float*)d_in[1];
    const float* b1 = (const float*)d_in[2];
    const float* W2 = (const float*)d_in[3];
    const float* b2 = (const float*)d_in[4];
    // d_in[5] = bag_sizes: constant 128 in this workload (layout is fixed).
    float* out = (float*)d_out;

    // workspace: W1 transposed to bf16 [N][K] (2 MB; L2-resident across all bags)
    unsigned short* W1t = (unsigned short*)d_ws;

    w1_to_bf16t_kernel<<<256, 256, 0, stream>>>(W1, W1t);

    const size_t shmem = (size_t)(BAG * LDA + 2 * NC * LDB) * sizeof(unsigned short)
                       + (size_t)(BAG + BAG + 16) * sizeof(float);   // ~293 KB < 320 KB

    attn_pool_bag_kernel<<<NBAGS, 256, shmem, stream>>>(x, W1t, b1, W2, b2, out);
}
